// TorsionLoss_82051055223045
// MI455X (gfx1250) — compile-verified, matrix operand order checked by reading, not run on hardware
//
#include <hip/hip_runtime.h>
#include <hip/hip_bf16.h>
#include <math.h>

typedef __bf16 bf16_t;
typedef __attribute__((ext_vector_type(16))) __bf16 v16bf;
typedef __attribute__((ext_vector_type(8)))  __bf16 v8bf;
typedef __attribute__((ext_vector_type(8)))  float  v8f;

#define HID   512
#define BATCH 32768
#define KAPPA (1.0f / 61.0f)
#define EPS   1e-10f
#define LDSE  520   // padded LDS row (elements): 1040B = 260 dwords == 4 (mod 64)
                    // -> lane n's fragment starts at bank 4n: conflict-free b128 reads

// Workspace layout:
//   [0   ..   8) : float acc[2]  (loss-sum, norm-sum)
//   [256 .. 256+512*512*2) : bf16 G[512][512]  (dense, row-major)

// ---------------------------------------------------------------------------
// Kernel 1: G[j][jp] = (W1^T W1)[j,jp] * (W2 W2^T)[j,jp], stored bf16.
// Also zeroes the two global accumulators (stream-ordered before kernel 2).
// ---------------------------------------------------------------------------
__global__ void torsion_prepare_G(const float* __restrict__ W1,
                                  const float* __restrict__ W2,
                                  bf16_t* __restrict__ G,
                                  float* __restrict__ acc) {
    int idx = blockIdx.x * blockDim.x + threadIdx.x;
    if (idx < 2) acc[idx] = 0.0f;
    if (idx >= HID * HID) return;
    int j  = idx >> 9;
    int jp = idx & 511;
    float d1 = 0.0f;
#pragma unroll
    for (int i = 0; i < 7; ++i)
        d1 += W1[i * HID + j] * W1[i * HID + jp];
    float d2 = 0.0f;
#pragma unroll
    for (int m = 0; m < 35; ++m)
        d2 += W2[j * 35 + m] * W2[jp * 35 + m];
    G[idx] = (bf16_t)(d1 * d2);
}

// ---------------------------------------------------------------------------
// Async copy of one 16x512 bf16 G tile (16KB) into a padded LDS tile.
// 256 threads x 4 chunks x 16B. gfx1250 async path: tracked by ASYNCcnt.
// Low 32 bits of a generic pointer to __shared__ == LDS byte address
// (flat LDS aperture truncation, ISA 10.2).
// ---------------------------------------------------------------------------
__device__ __forceinline__ void async_copy_tile(const bf16_t* __restrict__ gTile,
                                                bf16_t* ldsTile, int tid) {
#pragma unroll
    for (int i = 0; i < 4; ++i) {
        const int chunk = tid + i * 256;            // 16B chunk id, 0..1023
        const int r     = chunk >> 6;               // tile row 0..15
        const int cb    = (chunk & 63) << 4;        // byte within row
        unsigned int ldsAddr = (unsigned int)(uintptr_t)ldsTile
                             + (unsigned int)(r * (LDSE * 2) + cb);
        unsigned int vOff    = (unsigned int)(r * (HID * 2) + cb);
        unsigned long long sBase = (unsigned long long)(uintptr_t)gTile;
        asm volatile("global_load_async_to_lds_b128 %0, %1, %2"
                     :: "v"(ldsAddr), "v"(vOff), "s"(sBase)
                     : "memory");
    }
}

// 32B fragment as two 16B (b128) LDS loads; 1040B row stride is 16B-aligned.
__device__ __forceinline__ v16bf load_frag(const bf16_t* p) {
    union { v16bf v; v8bf h[2]; } f;
    f.h[0] = *(const v8bf*)(p);
    f.h[1] = *(const v8bf*)(p + 8);
    return f.v;
}

// ---------------------------------------------------------------------------
// Kernel 2: per wave: 16 rows.  u = 1 - tanh(xW1+b1)^2 into LDS (bf16),
// then t_b = 6 * u^T G u via v_wmma_f32_16x16x32_bf16.  G N-tiles are staged
// block-wide in LDS via triple-buffered async loads (ASYNCcnt).
// Block = 256 threads = 8 waves -> 128 rows/block, 256 blocks.
// ---------------------------------------------------------------------------
__global__ __launch_bounds__(256, 1)
void torsion_main_kernel(const float* __restrict__ x,
                         const float* __restrict__ W1,
                         const float* __restrict__ b1,
                         const bf16_t* __restrict__ G,
                         float* __restrict__ acc) {
    __shared__ bf16_t ldsU[8][16][LDSE];   // 8 waves * 16 rows : ~130 KB
    __shared__ bf16_t ldsB[3][16][LDSE];   // triple-buffered G tile : ~49 KB

    const int tid     = threadIdx.x;
    const int lane    = tid & 31;
    const int wave    = tid >> 5;
    const int rowBase = blockIdx.x * 128 + wave * 16;

    // Kick off tile 0 copy immediately; it overlaps the whole U phase.
    async_copy_tile(G, &ldsB[0][0][0], tid);

    // ---- Phase 1: compute this wave's 16 rows of U into LDS ----
    for (int r = 0; r < 16; ++r) {
        const float* xr = x + (size_t)(rowBase + r) * 7;
        const float x0 = xr[0], x1 = xr[1], x2 = xr[2], x3 = xr[3];
        const float x4 = xr[4], x5 = xr[5], x6 = xr[6];
#pragma unroll 4
        for (int jj = 0; jj < 16; ++jj) {
            const int j = jj * 32 + lane;           // coalesced across lanes
            float z = b1[j]
                    + x0 * W1[0 * HID + j] + x1 * W1[1 * HID + j]
                    + x2 * W1[2 * HID + j] + x3 * W1[3 * HID + j]
                    + x4 * W1[4 * HID + j] + x5 * W1[5 * HID + j]
                    + x6 * W1[6 * HID + j];
            // sech^2(z) = 4 e^{2z} / (1 + e^{2z})^2  (one v_exp_f32)
            z = fminf(fmaxf(z, -15.0f), 15.0f);
            float e = __expf(2.0f * z);
            float s = 1.0f + e;
            ldsU[wave][r][j] = (bf16_t)(4.0f * e / (s * s));
        }
    }
    __syncthreads();

    // ---- Phase 2: GEMM-fused quadratic form ----
    const int half = lane >> 4;     // K-half selector (A/B fragment layout)
    const int l16  = lane & 15;     // M row (A) / N column (B) / N (C,D)

    // Hoist all 16 A fragments (row = l16, contiguous K) into VGPRs: 128 VGPRs.
    const bf16_t* uRow = &ldsU[wave][l16][half * 16];
    v16bf afrag[16];
#pragma unroll
    for (int kt = 0; kt < 16; ++kt)
        afrag[kt] = load_frag(uRow + kt * 32);

    float ssum[8];
#pragma unroll
    for (int v = 0; v < 8; ++v) ssum[v] = 0.0f;

    for (int nt = 0; nt < 32; ++nt) {
        // Prefetch tile nt+1 into buffer (nt+1)%3; wait for tile nt's 4 chunks.
        // Triple buffering + the barrier below covers the WAR hazard.
        if (nt + 1 < 32) {
            async_copy_tile(G + (size_t)(nt + 1) * 16 * HID,
                            &ldsB[(nt + 1) % 3][0][0], tid);
            asm volatile("s_wait_asynccnt 4" ::: "memory");
        } else {
            asm volatile("s_wait_asynccnt 0" ::: "memory");
        }
        __syncthreads();

        // B fragment: column n = l16 of tile == row (n0+l16) of G (G symmetric),
        // now a conflict-free LDS read.
        const bf16_t* bRow = &ldsB[nt % 3][l16][half * 16];

        v8f c = {0.f, 0.f, 0.f, 0.f, 0.f, 0.f, 0.f, 0.f};
#pragma unroll
        for (int kt = 0; kt < 16; ++kt) {
            v16bf b = load_frag(bRow + kt * 32);
            c = __builtin_amdgcn_wmma_f32_16x16x32_bf16(
                    false, afrag[kt], false, b, (short)0, c, false, false);
        }

        // Elementwise dot with U over this N tile.
        // C/D layout: VGPR v, lanes 0-15 -> (M=v, N=lane); lanes 16-31 -> (M=v+8).
        const int n0 = nt * 16;
#pragma unroll
        for (int v = 0; v < 8; ++v) {
            const int m = v + 8 * half;
            ssum[v] += c[v] * (float)ldsU[wave][m][n0 + l16];
        }
    }

    // ---- Phase 3: reduce over N (16 lanes per half), then loss terms ----
#pragma unroll
    for (int v = 0; v < 8; ++v) {
        ssum[v] += __shfl_xor(ssum[v], 1);
        ssum[v] += __shfl_xor(ssum[v], 2);
        ssum[v] += __shfl_xor(ssum[v], 4);
        ssum[v] += __shfl_xor(ssum[v], 8);
    }

    if (l16 == 0) {   // lane 0 handles rows 0..7, lane 16 handles rows 8..15
        float lossAcc = 0.0f, normAcc = 0.0f;
#pragma unroll
        for (int v = 0; v < 8; ++v) {
            float tsq = 6.0f * ssum[v];
            float nrm = sqrtf(tsq + EPS);
            float d   = nrm - KAPPA;
            lossAcc += d * d;
            normAcc += nrm;
        }
        atomicAdd(&acc[0], lossAcc);
        atomicAdd(&acc[1], normAcc);
    }
}

// ---------------------------------------------------------------------------
// Kernel 3: finalize means.
// ---------------------------------------------------------------------------
__global__ void torsion_finalize(const float* __restrict__ acc,
                                 float* __restrict__ out) {
    if (threadIdx.x == 0) {
        out[0] = acc[0] * (1.0f / (float)BATCH);
        out[1] = acc[1] * (1.0f / (float)BATCH);
    }
}

extern "C" void kernel_launch(void* const* d_in, const int* in_sizes, int n_in,
                              void* d_out, int out_size, void* d_ws, size_t ws_size,
                              hipStream_t stream) {
    const float* x  = (const float*)d_in[0];   // (32768, 7)
    const float* W1 = (const float*)d_in[1];   // (7, 512)
    const float* b1 = (const float*)d_in[2];   // (512,)
    const float* W2 = (const float*)d_in[3];   // (512, 35)
    // d_in[4] (b2) is mathematically irrelevant: constant offset vanishes in Jacobian.

    float*  acc = (float*)d_ws;
    bf16_t* G   = (bf16_t*)((char*)d_ws + 256);
    float*  out = (float*)d_out;

    torsion_prepare_G<<<(HID * HID + 255) / 256, 256, 0, stream>>>(W1, W2, G, acc);
    torsion_main_kernel<<<BATCH / 128, 256, 0, stream>>>(x, W1, b1, G, acc);
    torsion_finalize<<<1, 32, 0, stream>>>(acc, out);
}
